// VotingDynamicsEngine_37752762532575
// MI455X (gfx1250) — compile-verified, matrix operand order checked by reading, not run on hardware
//
#include <hip/hip_runtime.h>
#include <math.h>

typedef __attribute__((ext_vector_type(16))) __bf16 bf16x16;
typedef __attribute__((ext_vector_type(8)))  __bf16 bf16x8;
typedef __attribute__((ext_vector_type(8)))  float  f32x8;

#define N_CELLS 4096
#define IN_DIM  1024
#define HID     1024
#define OUT_DIM 1024
#define MID     128
#define NT      8
#define NF      8
#define CDIM    2048          // IN_DIM + HID
#define GIN     1025          // OUT_DIM + 1
#define GINP    1056          // GIN padded to multiple of 32
#define FS      (N_CELLS / NF)  // 512
#define DC      (FS / 4)        // 128

// ---------------------------------------------------------------- WMMA helpers
static __device__ __forceinline__ f32x8 wmma_bf16(bf16x16 a, bf16x16 b, f32x8 c) {
  return __builtin_amdgcn_wmma_f32_16x16x32_bf16(false, a, false, b, (short)0, c,
                                                 false, false);
}

// Per-lane fragment load for 16x16x32 bf16 WMMA.
// Caller passes p = base + row*stride + k0 where row = Tb + (lane&15),
// k0 = kk + ((lane>>4)<<3).  Chunks: [k0..k0+7] and [k0+16..k0+23].
static __device__ __forceinline__ bf16x16 load_frag(const __bf16* p) {
  bf16x8 lo = *(const bf16x8*)(p);
  bf16x8 hi = *(const bf16x8*)(p + 16);
  bf16x16 r;
#pragma unroll
  for (int i = 0; i < 8; ++i) { r[i] = lo[i]; r[i + 8] = hi[i]; }
  return r;
}

// ---------------------------------------------------------------- pack kernels
__global__ void k_pack_comb(const float* __restrict__ x, const float* __restrict__ hid,
                            __bf16* __restrict__ comb) {
  int idx = blockIdx.x * 256 + threadIdx.x;
  if (idx >= N_CELLS * CDIM) return;
  int i = idx >> 11, j = idx & (CDIM - 1);
  float v = (j < IN_DIM) ? x[j] : hid[(size_t)i * HID + (j - IN_DIM)];
  comb[idx] = (__bf16)v;
}

__global__ void k_cvt_bf16(const float* __restrict__ s, __bf16* __restrict__ d, int n) {
  int idx = blockIdx.x * 256 + threadIdx.x;
  if (idx < n) d[idx] = (__bf16)s[idx];
}

__global__ void k_pack_w1(const float* __restrict__ Wa1, const float* __restrict__ Wg1,
                          const float* __restrict__ ba1, const float* __restrict__ bg1,
                          __bf16* __restrict__ w1, float* __restrict__ b1) {
  int idx = blockIdx.x * 256 + threadIdx.x;       // 256*2048
  if (idx >= 256 * CDIM) return;
  int r = idx >> 11, j = idx & (CDIM - 1);
  float v = (r < MID) ? Wa1[(size_t)r * CDIM + j] : Wg1[(size_t)(r - MID) * CDIM + j];
  w1[idx] = (__bf16)v;
  if (j == 0) b1[r] = (r < MID) ? ba1[r] : bg1[r - MID];
}

__global__ void k_pack_w2(const float* __restrict__ Wa2, const float* __restrict__ Wg2,
                          const float* __restrict__ ba2, const float* __restrict__ bg2,
                          __bf16* __restrict__ w2, float* __restrict__ b2) {
  int idx = blockIdx.x * 256 + threadIdx.x;       // 1024*256
  if (idx >= OUT_DIM * 256) return;
  int o = idx >> 8, m = idx & 255;
  float v = (m < MID) ? Wa2[(size_t)o * MID + m] : -Wg2[(size_t)o * MID + (m - MID)];
  w2[idx] = (__bf16)v;
  if (m == 0) b2[o] = ba2[o] - bg2[o];
}

__global__ void k_pack_wih(const float* __restrict__ Wih, __bf16* __restrict__ w) {
  int idx = blockIdx.x * 256 + threadIdx.x;       // 3072*1056
  if (idx >= 3 * HID * GINP) return;
  int r = idx / GINP, j = idx % GINP;
  float v = (j < GIN) ? Wih[(size_t)r * GIN + j] : 0.f;
  w[idx] = (__bf16)v;
}

__global__ void k_zero(float* p, int n) {
  int i = blockIdx.x * 256 + threadIdx.x;
  if (i < n) p[i] = 0.f;
}

// ---------------------------------------------------------------- vote round
__global__ void k_vote(const float* __restrict__ infl, const float* __restrict__ op,
                       const float* __restrict__ noise, int* __restrict__ votes,
                       float* __restrict__ energy) {
  int row = blockIdx.x, t = threadIdx.x;          // 256 threads
  float acc[NT] = {};
  const float* irow = infl + (size_t)row * N_CELLS;
  for (int j = t; j < N_CELLS; j += 256) {
    float wv = irow[j];
    if (wv != 0.f) {                              // influence is ~94% zeros
      const float* o = op + (size_t)j * NT;
#pragma unroll
      for (int k = 0; k < NT; ++k) acc[k] += wv * o[k];
    }
  }
  __shared__ float red[NT * 256];
#pragma unroll
  for (int k = 0; k < NT; ++k) red[k * 256 + t] = acc[k];
  __syncthreads();
  for (int s = 128; s > 0; s >>= 1) {
    if (t < s) {
#pragma unroll
      for (int k = 0; k < NT; ++k) red[k * 256 + t] += red[k * 256 + t + s];
    }
    __syncthreads();
  }
  if (t == 0) {
    float no[NT]; float sum = 0.f;
#pragma unroll
    for (int k = 0; k < NT; ++k) {
      float v = 0.7f * op[(size_t)row * NT + k] + 0.3f * red[k * 256] +
                0.01f * noise[(size_t)row * NT + k];
      v = fmaxf(v, 0.01f);
      no[k] = v; sum += v;
    }
    float inv = 1.f / sum, best = -1.f; int bi = 0;
#pragma unroll
    for (int k = 0; k < NT; ++k) {
      float v = no[k] * inv;
      if (v > best) { best = v; bi = k; }
    }
    votes[row] = bi;
    energy[row] = best;
  }
}

// -------------------------------------------------- GEMM1: comb @ [Wa1|Wg1]^T, relu
// 1 M-tile x 4 N-tiles per wave, software-pipelined (double-buffered fragments).
__global__ void k_gemm1(const __bf16* __restrict__ A, const __bf16* __restrict__ B,
                        const float* __restrict__ b1, __bf16* __restrict__ mid) {
  int wid  = (blockIdx.x * blockDim.x + threadIdx.x) >> 5;  // 1024 waves
  int lane = threadIdx.x & 31;
  int Mb  = (wid >> 2) << 4;                   // 256 M tiles
  int Nb0 = (wid & 3) << 6;                    // 4 N groups of 64 (N=256)
  int arow = Mb + (lane & 15);
  int ksel = (lane >> 4) << 3;
  const __bf16* ap = A + (size_t)arow * CDIM + ksel;
  const __bf16* bp[4];
#pragma unroll
  for (int j = 0; j < 4; ++j)
    bp[j] = B + (size_t)(Nb0 + j * 16 + (lane & 15)) * CDIM + ksel;
  f32x8 acc[4];
#pragma unroll
  for (int j = 0; j < 4; ++j) acc[j] = (f32x8){0.f,0.f,0.f,0.f,0.f,0.f,0.f,0.f};

  bf16x16 a = load_frag(ap);
  bf16x16 b[4];
#pragma unroll
  for (int j = 0; j < 4; ++j) b[j] = load_frag(bp[j]);

  for (int kk = 32; kk < CDIM; kk += 32) {
    // issue next iteration's loads before consuming current fragments
    bf16x16 an = load_frag(ap + kk);
    bf16x16 bn[4];
#pragma unroll
    for (int j = 0; j < 4; ++j) bn[j] = load_frag(bp[j] + kk);
#pragma unroll
    for (int j = 0; j < 4; ++j) acc[j] = wmma_bf16(a, b[j], acc[j]);
    a = an;
#pragma unroll
    for (int j = 0; j < 4; ++j) b[j] = bn[j];
  }
#pragma unroll
  for (int j = 0; j < 4; ++j) acc[j] = wmma_bf16(a, b[j], acc[j]);

  int m0 = Mb + ((lane >> 4) << 3);
#pragma unroll
  for (int j = 0; j < 4; ++j) {
    int n = Nb0 + j * 16 + (lane & 15);
    float bias = b1[n];
#pragma unroll
    for (int r = 0; r < 8; ++r) {
      float v = acc[j][r] + bias;
      mid[(size_t)(m0 + r) * 256 + n] = (__bf16)(v > 0.f ? v : 0.f);
    }
  }
}

// -------------------------------------------------- GEMM2: mid @ [Wa2|-Wg2]^T -> out (f32)
// 1 M-tile x 4 N-tiles per wave, software-pipelined.
__global__ void k_gemm2(const __bf16* __restrict__ A, const __bf16* __restrict__ B,
                        const float* __restrict__ b2, float* __restrict__ out) {
  int wid  = (blockIdx.x * blockDim.x + threadIdx.x) >> 5;  // 4096 waves
  int lane = threadIdx.x & 31;
  int Mb  = (wid >> 4) << 4;                   // 256 M tiles
  int Nb0 = (wid & 15) << 6;                   // 16 N groups of 64 (N=1024)
  int arow = Mb + (lane & 15);
  int ksel = (lane >> 4) << 3;
  const __bf16* ap = A + (size_t)arow * 256 + ksel;
  const __bf16* bp[4];
#pragma unroll
  for (int j = 0; j < 4; ++j)
    bp[j] = B + (size_t)(Nb0 + j * 16 + (lane & 15)) * 256 + ksel;
  f32x8 acc[4];
#pragma unroll
  for (int j = 0; j < 4; ++j) acc[j] = (f32x8){0.f,0.f,0.f,0.f,0.f,0.f,0.f,0.f};

  bf16x16 a = load_frag(ap);
  bf16x16 b[4];
#pragma unroll
  for (int j = 0; j < 4; ++j) b[j] = load_frag(bp[j]);

  for (int kk = 32; kk < 256; kk += 32) {
    bf16x16 an = load_frag(ap + kk);
    bf16x16 bn[4];
#pragma unroll
    for (int j = 0; j < 4; ++j) bn[j] = load_frag(bp[j] + kk);
#pragma unroll
    for (int j = 0; j < 4; ++j) acc[j] = wmma_bf16(a, b[j], acc[j]);
    a = an;
#pragma unroll
    for (int j = 0; j < 4; ++j) b[j] = bn[j];
  }
#pragma unroll
  for (int j = 0; j < 4; ++j) acc[j] = wmma_bf16(a, b[j], acc[j]);

  int m0 = Mb + ((lane >> 4) << 3);
#pragma unroll
  for (int j = 0; j < 4; ++j) {
    int n = Nb0 + j * 16 + (lane & 15);
    float bias = b2[n];
#pragma unroll
    for (int r = 0; r < 8; ++r)
      out[(size_t)(m0 + r) * OUT_DIM + n] = acc[j][r] + bias;
  }
}

// ---------------------------------------------------------------- tension + mem_in pack
__global__ void k_tension(const float* __restrict__ out, float* __restrict__ tension,
                          __bf16* __restrict__ memin) {
  int row = blockIdx.x, t = threadIdx.x;          // 256 threads
  float v[4]; float s = 0.f;
#pragma unroll
  for (int i = 0; i < 4; ++i) {
    v[i] = out[(size_t)row * OUT_DIM + t + i * 256];
    s += v[i] * v[i];
  }
  __shared__ float sh[256];
  sh[t] = s; __syncthreads();
  for (int st = 128; st > 0; st >>= 1) {
    if (t < st) sh[t] += sh[t + st];
    __syncthreads();
  }
#pragma unroll
  for (int i = 0; i < 4; ++i)
    memin[(size_t)row * GINP + t + i * 256] = (__bf16)v[i];
  if (t == 0) {
    float tv = sh[0] * (1.f / OUT_DIM);
    tension[row] = tv;
    memin[(size_t)row * GINP + OUT_DIM] = (__bf16)tv;
    for (int c = GIN; c < GINP; ++c) memin[(size_t)row * GINP + c] = (__bf16)0.f;
  }
}

// -------------------------------------------------- fused GRU: 2 M-tiles x 3 gates x 2 GEMMs
// 12 WMMA accumulators; fragments double-buffered so 5 loads overlap 6 WMMAs.
__global__ void k_gru(const __bf16* __restrict__ Amem, const __bf16* __restrict__ Wih,
                      const __bf16* __restrict__ Ahid, const __bf16* __restrict__ Whh,
                      const float* __restrict__ b_ih, const float* __restrict__ b_hh,
                      const float* __restrict__ hiddens, const float* __restrict__ energy,
                      float* __restrict__ newh) {
  int wid  = (blockIdx.x * blockDim.x + threadIdx.x) >> 5;  // 8192 waves
  int lane = threadIdx.x & 31;
  int Mb = (wid >> 6) << 5;                    // 128 M blocks of 32 rows
  int Nb = (wid & 63) << 4;                    // 64 N tiles (HID=1024)
  int arow0 = Mb + (lane & 15);
  int bn    = Nb + (lane & 15);
  int ksel  = (lane >> 4) << 3;

  f32x8 z8 = {0.f,0.f,0.f,0.f,0.f,0.f,0.f,0.f};
  f32x8 gi[3][2] = {{z8, z8}, {z8, z8}, {z8, z8}};
  f32x8 gh[3][2] = {{z8, z8}, {z8, z8}, {z8, z8}};

  // gi = mem_in @ W_ih^T  (K = 1056, zero-padded)
  {
    const __bf16* ap0 = Amem + (size_t)arow0 * GINP + ksel;
    const __bf16* ap1 = ap0 + (size_t)16 * GINP;
    const __bf16* bp[3];
#pragma unroll
    for (int g = 0; g < 3; ++g)
      bp[g] = Wih + (size_t)(bn + g * HID) * GINP + ksel;

    bf16x16 a0 = load_frag(ap0), a1 = load_frag(ap1);
    bf16x16 b[3];
#pragma unroll
    for (int g = 0; g < 3; ++g) b[g] = load_frag(bp[g]);

    for (int kk = 32; kk < GINP; kk += 32) {
      bf16x16 a0n = load_frag(ap0 + kk);
      bf16x16 a1n = load_frag(ap1 + kk);
      bf16x16 bnx[3];
#pragma unroll
      for (int g = 0; g < 3; ++g) bnx[g] = load_frag(bp[g] + kk);
#pragma unroll
      for (int g = 0; g < 3; ++g) {
        gi[g][0] = wmma_bf16(a0, b[g], gi[g][0]);
        gi[g][1] = wmma_bf16(a1, b[g], gi[g][1]);
      }
      a0 = a0n; a1 = a1n;
#pragma unroll
      for (int g = 0; g < 3; ++g) b[g] = bnx[g];
    }
#pragma unroll
    for (int g = 0; g < 3; ++g) {
      gi[g][0] = wmma_bf16(a0, b[g], gi[g][0]);
      gi[g][1] = wmma_bf16(a1, b[g], gi[g][1]);
    }
  }
  // gh = hiddens @ W_hh^T  (K = 1024)
  {
    const __bf16* ap0 = Ahid + (size_t)arow0 * HID + ksel;
    const __bf16* ap1 = ap0 + (size_t)16 * HID;
    const __bf16* bp[3];
#pragma unroll
    for (int g = 0; g < 3; ++g)
      bp[g] = Whh + (size_t)(bn + g * HID) * HID + ksel;

    bf16x16 a0 = load_frag(ap0), a1 = load_frag(ap1);
    bf16x16 b[3];
#pragma unroll
    for (int g = 0; g < 3; ++g) b[g] = load_frag(bp[g]);

    for (int kk = 32; kk < HID; kk += 32) {
      bf16x16 a0n = load_frag(ap0 + kk);
      bf16x16 a1n = load_frag(ap1 + kk);
      bf16x16 bnx[3];
#pragma unroll
      for (int g = 0; g < 3; ++g) bnx[g] = load_frag(bp[g] + kk);
#pragma unroll
      for (int g = 0; g < 3; ++g) {
        gh[g][0] = wmma_bf16(a0, b[g], gh[g][0]);
        gh[g][1] = wmma_bf16(a1, b[g], gh[g][1]);
      }
      a0 = a0n; a1 = a1n;
#pragma unroll
      for (int g = 0; g < 3; ++g) b[g] = bnx[g];
    }
#pragma unroll
    for (int g = 0; g < 3; ++g) {
      gh[g][0] = wmma_bf16(a0, b[g], gh[g][0]);
      gh[g][1] = wmma_bf16(a1, b[g], gh[g][1]);
    }
  }
  // epilogue: full GRU nonlinearity + energy scale + clip
  int n = Nb + (lane & 15);
  float bir = b_ih[n], biz = b_ih[HID + n], bin = b_ih[2 * HID + n];
  float bhr = b_hh[n], bhz = b_hh[HID + n], bhn = b_hh[2 * HID + n];
#pragma unroll
  for (int mt = 0; mt < 2; ++mt) {
    int m0 = Mb + mt * 16 + ((lane >> 4) << 3);
#pragma unroll
    for (int r = 0; r < 8; ++r) {
      int m = m0 + r;
      float rg = 1.f / (1.f + __expf(-(gi[0][mt][r] + bir + gh[0][mt][r] + bhr)));
      float zg = 1.f / (1.f + __expf(-(gi[1][mt][r] + biz + gh[1][mt][r] + bhz)));
      float ng = tanhf(gi[2][mt][r] + bin + rg * (gh[2][mt][r] + bhn));
      float h  = hiddens[(size_t)m * HID + n];
      float nh = (1.f - zg) * ng + zg * h;
      nh *= (0.9f + 0.2f * energy[m]);
      nh = fminf(10.f, fmaxf(-10.f, nh));
      newh[(size_t)m * HID + n] = nh;
    }
  }
}

// ---------------------------------------------------------------- topic segment sums / counts / blend
__global__ void k_segsum(const float* __restrict__ nh, const int* __restrict__ votes,
                         float* __restrict__ sums) {
  __shared__ float loc[NT][256];
  int t = threadIdx.x;
#pragma unroll
  for (int f = 0; f < NT; ++f) loc[f][t] = 0.f;
  __syncthreads();
  int c  = (blockIdx.x & 3) * 256 + t;            // 4 column chunks
  int r0 = (blockIdx.x >> 2) * 128;               // 32 row chunks
  for (int r = r0; r < r0 + 128; ++r)
    loc[votes[r]][t] += nh[(size_t)r * HID + c];
  __syncthreads();
#pragma unroll
  for (int f = 0; f < NT; ++f) atomicAdd(&sums[(size_t)f * HID + c], loc[f][t]);
}

__global__ void k_count(const int* __restrict__ votes, float* __restrict__ counts) {
  __shared__ int lc[NT];
  if (threadIdx.x < NT) lc[threadIdx.x] = 0;
  __syncthreads();
  for (int i = threadIdx.x; i < N_CELLS; i += 256) atomicAdd(&lc[votes[i]], 1);
  __syncthreads();
  if (threadIdx.x < NT) counts[threadIdx.x] = (float)lc[threadIdx.x];
}

__global__ void k_tblend(float* __restrict__ nh, const int* __restrict__ votes,
                         const float* __restrict__ sums, const float* __restrict__ counts) {
  size_t idx = (size_t)blockIdx.x * 256 + threadIdx.x;
  int i = (int)(idx >> 10), c = (int)(idx & 1023);
  int v = votes[i];
  float cnt = counts[v];
  if (cnt >= 2.f) {
    float mean = sums[(size_t)v * HID + c] / fmaxf(cnt, 1.f);
    nh[idx] = 0.85f * nh[idx] + 0.15f * mean;
  }
}

// ---------------------------------------------------------------- faction sync
__global__ void k_fmean_part(const float* __restrict__ nh, float* __restrict__ fm) {
  int bx = blockIdx.x;                            // 8 f * 4 colchunk * 8 rowchunk = 256
  int f  = bx >> 5;
  int cc = (bx >> 3) & 3;
  int rc = bx & 7;
  int c  = cc * 256 + threadIdx.x;
  int r0 = f * FS + rc * 64;
  float s = 0.f;
  for (int r = r0; r < r0 + 64; ++r) s += nh[(size_t)r * HID + c];
  atomicAdd(&fm[(size_t)f * HID + c], s);
}

__global__ void k_glob(float* __restrict__ fm, float* __restrict__ glob) {
  int c = blockIdx.x * 256 + threadIdx.x;         // 1024
  float g = 0.f;
#pragma unroll
  for (int f = 0; f < NF; ++f) {
    float v = fm[(size_t)f * HID + c] * (1.f / FS);
    fm[(size_t)f * HID + c] = v;
    g += v;
  }
  glob[c] = g * (1.f / NF);
}

__global__ void k_fblend(float* __restrict__ nh, const float* __restrict__ fm,
                         const float* __restrict__ glob, const int* __restrict__ step) {
  size_t idx = (size_t)blockIdx.x * 256 + threadIdx.x;
  int i = (int)(idx >> 10), c = (int)(idx & 1023);
  int f = i >> 9, ric = i & (FS - 1);
  float v = nh[idx];
  v = 0.85f * v + 0.15f * fm[(size_t)f * HID + c];
  if (ric < DC && step[0] > 5) v = 0.85f * v + 0.15f * glob[c];
  nh[idx] = v;
}

// ---------------------------------------------------------------- softmax(tension), combined, pred
__global__ void k_softmax(const float* __restrict__ tension, float* __restrict__ w,
                          float* __restrict__ avg_out) {
  __shared__ float sh[256];
  int t = threadIdx.x;
  float tv[16]; float mx = -1e30f, tsum = 0.f;
#pragma unroll
  for (int i = 0; i < 16; ++i) {
    tv[i] = tension[t + 256 * i];
    mx = fmaxf(mx, tv[i]);
    tsum += tv[i];
  }
  sh[t] = mx; __syncthreads();
  for (int s = 128; s > 0; s >>= 1) { if (t < s) sh[t] = fmaxf(sh[t], sh[t + s]); __syncthreads(); }
  float tmax = sh[0]; __syncthreads();
  sh[t] = tsum; __syncthreads();
  for (int s = 128; s > 0; s >>= 1) { if (t < s) sh[t] += sh[t + s]; __syncthreads(); }
  float total = sh[0]; __syncthreads();
  float es = 0.f;
#pragma unroll
  for (int i = 0; i < 16; ++i) { tv[i] = __expf(tv[i] - tmax); es += tv[i]; }
  sh[t] = es; __syncthreads();
  for (int s = 128; s > 0; s >>= 1) { if (t < s) sh[t] += sh[t + s]; __syncthreads(); }
  float inv = 1.f / sh[0];
#pragma unroll
  for (int i = 0; i < 16; ++i) w[t + 256 * i] = tv[i] * inv;
  if (t == 0) avg_out[0] = total * (1.f / N_CELLS);
}

__global__ void k_combined(const float* __restrict__ w, const float* __restrict__ out,
                           float* __restrict__ cmb) {
  int c = blockIdx.x * 256 + threadIdx.x;         // 1024
  float s = 0.f;
  for (int i = 0; i < N_CELLS; ++i) s += w[i] * out[(size_t)i * OUT_DIM + c];
  cmb[c] = s;
}

__global__ void k_pred(const float* __restrict__ cmb, const float* __restrict__ Wo,
                       const float* __restrict__ bo, float* __restrict__ pred) {
  int j = blockIdx.x, t = threadIdx.x;            // 1024 blocks x 128 threads
  __shared__ float sh[128];
  float s = 0.f;
  for (int o = t; o < OUT_DIM; o += 128) s += cmb[o] * Wo[(size_t)j * OUT_DIM + o];
  sh[t] = s; __syncthreads();
  for (int st = 64; st > 0; st >>= 1) { if (t < st) sh[t] += sh[t + st]; __syncthreads(); }
  if (t == 0) pred[j] = sh[0] + bo[j];
}

// ---------------------------------------------------------------- launch
extern "C" void kernel_launch(void* const* d_in, const int* in_sizes, int n_in,
                              void* d_out, int out_size, void* d_ws, size_t ws_size,
                              hipStream_t stream) {
  (void)in_sizes; (void)n_in; (void)out_size; (void)ws_size;
  const float* x        = (const float*)d_in[0];
  const float* hiddens  = (const float*)d_in[1];
  const float* opinions = (const float*)d_in[2];
  const float* influence= (const float*)d_in[3];
  const float* vnoise   = (const float*)d_in[4];
  const float* Wa1 = (const float*)d_in[5];  const float* ba1 = (const float*)d_in[6];
  const float* Wa2 = (const float*)d_in[7];  const float* ba2 = (const float*)d_in[8];
  const float* Wg1 = (const float*)d_in[9];  const float* bg1 = (const float*)d_in[10];
  const float* Wg2 = (const float*)d_in[11]; const float* bg2 = (const float*)d_in[12];
  const float* W_ih = (const float*)d_in[13]; const float* b_ih = (const float*)d_in[14];
  const float* W_hh = (const float*)d_in[15]; const float* b_hh = (const float*)d_in[16];
  const float* Wo  = (const float*)d_in[17]; const float* bo  = (const float*)d_in[18];
  const int*   step = (const int*)d_in[19];

  float* out_pred = (float*)d_out;               // [1024]
  float* out_avg  = (float*)d_out + 1024;        // [1]
  float* newh     = (float*)d_out + 1025;        // [4096*1024]

  char* cur = (char*)d_ws;
  auto alloc = [&](size_t bytes) -> void* {
    void* p = cur;
    cur += (bytes + 255) & ~(size_t)255;
    return p;
  };
  __bf16* comb  = (__bf16*)alloc((size_t)N_CELLS * CDIM * 2);
  __bf16* w1    = (__bf16*)alloc((size_t)256 * CDIM * 2);
  float*  b1    = (float*)alloc(256 * 4);
  __bf16* midb  = (__bf16*)alloc((size_t)N_CELLS * 256 * 2);
  __bf16* w2    = (__bf16*)alloc((size_t)OUT_DIM * 256 * 2);
  float*  b2    = (float*)alloc(OUT_DIM * 4);
  float*  outm  = (float*)alloc((size_t)N_CELLS * OUT_DIM * 4);
  float*  tens  = (float*)alloc(N_CELLS * 4);
  __bf16* memin = (__bf16*)alloc((size_t)N_CELLS * GINP * 2);
  __bf16* wihb  = (__bf16*)alloc((size_t)3 * HID * GINP * 2);
  __bf16* hidb  = (__bf16*)alloc((size_t)N_CELLS * HID * 2);
  __bf16* whhb  = (__bf16*)alloc((size_t)3 * HID * HID * 2);
  int*    votes = (int*)alloc(N_CELLS * 4);
  float*  energy= (float*)alloc(N_CELLS * 4);
  float*  sums  = (float*)alloc((size_t)NT * HID * 4);
  float*  counts= (float*)alloc(NT * 4);
  float*  fm    = (float*)alloc((size_t)NF * HID * 4);
  float*  glob  = (float*)alloc(HID * 4);
  float*  wsm   = (float*)alloc(N_CELLS * 4);
  float*  cmb   = (float*)alloc(OUT_DIM * 4);

  // --- packing / conversion
  k_pack_comb<<<(N_CELLS * CDIM) / 256, 256, 0, stream>>>(x, hiddens, comb);
  k_pack_w1<<<(256 * CDIM) / 256, 256, 0, stream>>>(Wa1, Wg1, ba1, bg1, w1, b1);
  k_pack_w2<<<(OUT_DIM * 256) / 256, 256, 0, stream>>>(Wa2, Wg2, ba2, bg2, w2, b2);
  k_pack_wih<<<(3 * HID * GINP + 255) / 256, 256, 0, stream>>>(W_ih, wihb);
  k_cvt_bf16<<<(N_CELLS * HID) / 256, 256, 0, stream>>>(hiddens, hidb, N_CELLS * HID);
  k_cvt_bf16<<<(3 * HID * HID) / 256, 256, 0, stream>>>(W_hh, whhb, 3 * HID * HID);

  // --- vote round
  k_vote<<<N_CELLS, 256, 0, stream>>>(influence, opinions, vnoise, votes, energy);

  // --- BenchMind GEMMs (WMMA, N-blocked x4, double-buffered)
  k_gemm1<<<128, 256, 0, stream>>>(comb, w1, b1, midb);       // 1024 waves
  k_gemm2<<<512, 256, 0, stream>>>(midb, w2, b2, outm);       // 4096 waves
  k_tension<<<N_CELLS, 256, 0, stream>>>(outm, tens, memin);

  // --- fused GRU GEMM (WMMA, M-blocked x2, double-buffered) -> new_h in d_out
  k_gru<<<1024, 256, 0, stream>>>(memin, wihb, hidb, whhb, b_ih, b_hh,
                                  hiddens, energy, newh);

  // --- topic-faction blending
  k_zero<<<(NT * HID + 255) / 256, 256, 0, stream>>>(sums, NT * HID);
  k_zero<<<1, 256, 0, stream>>>(counts, NT);
  k_zero<<<(NF * HID + 255) / 256, 256, 0, stream>>>(fm, NF * HID);
  k_segsum<<<128, 256, 0, stream>>>(newh, votes, sums);
  k_count<<<1, 256, 0, stream>>>(votes, counts);
  k_tblend<<<(N_CELLS * HID) / 256, 256, 0, stream>>>(newh, votes, sums, counts);

  // --- faction sync debate
  k_fmean_part<<<256, 256, 0, stream>>>(newh, fm);
  k_glob<<<4, 256, 0, stream>>>(fm, glob);
  k_fblend<<<(N_CELLS * HID) / 256, 256, 0, stream>>>(newh, fm, glob, step);

  // --- tension-weighted output
  k_softmax<<<1, 256, 0, stream>>>(tens, wsm, out_avg);
  k_combined<<<4, 256, 0, stream>>>(wsm, outm, cmb);
  k_pred<<<1024, 128, 0, stream>>>(cmb, Wo, bo, out_pred);
}
